// InterpretableMultiHeadAttention_84825604096154
// MI455X (gfx1250) — matrix operand order repaired and checked by reading, not compiled, not run on hardware
//
#include <hip/hip_runtime.h>

typedef __attribute__((ext_vector_type(16))) _Float16 v16h;
typedef __attribute__((ext_vector_type(8)))  _Float16 v8h;
typedef __attribute__((ext_vector_type(8)))  float    v8f;
typedef int gvec4i __attribute__((vector_size(16)));  // matches builtin's V4i pointee

namespace cfg {
constexpr int B = 4, S = 2048, D = 512, D3 = 1536;
constexpr long BS = (long)B * S;
}

#if defined(__has_builtin)
#if __has_builtin(__builtin_amdgcn_global_load_async_to_lds_b128)
#define USE_ASYNC_LDS 1
#endif
#endif
#ifndef USE_ASYNC_LDS
#define USE_ASYNC_LDS 0
#endif

// 16-byte global -> LDS copy. CDNA5 async path (ASYNCcnt, no VGPR round trip)
// when the builtin exists; plain load+ds_store otherwise.
static __device__ __forceinline__ void cp16_to_lds(const _Float16* g, _Float16* l) {
#if USE_ASYNC_LDS
  __builtin_amdgcn_global_load_async_to_lds_b128(
      (__attribute__((address_space(1))) gvec4i*)(void*)g,
      (__attribute__((address_space(3))) gvec4i*)(void*)l, 0, 0);
#else
  *(v8h*)l = *(const v8h*)g;
#endif
}

static __device__ __forceinline__ void wait_async_loads() {
#if USE_ASYNC_LDS
#if __has_builtin(__builtin_amdgcn_s_wait_asynccnt)
  __builtin_amdgcn_s_wait_asynccnt(0);
#else
  asm volatile("s_wait_asynccnt 0x0" ::: "memory");
#endif
  asm volatile("" ::: "memory");
#endif
}

static __device__ __forceinline__ v16h make_v16(v8h lo, v8h hi) {
  union { v16h v; v8h h[2]; } u;
  u.h[0] = lo; u.h[1] = hi;
  return u.v;
}

// ---------------------------------------------------------------------------
// Generic f16 WMMA GEMM:  C[M,N] = scale * (A[M,K] @ Bt[N,K]^T) + bias[n]
// A row-major [M x K] (leading dim lda), Bt row-major [N x K] (ldb).
// Block tile 128x128, 8 waves in 4(m) x 2(n), each wave: 2x4 16x16 tiles.
// K staged through double-buffered LDS in 64-wide slabs; next slab is
// prefetched (async-to-LDS on CDNA5) while the current slab feeds the WMMAs.
// All dims assumed multiples of the tile sizes (true for every stage here).
// ---------------------------------------------------------------------------
template <bool OUT_F16, bool HAS_BIAS>
__global__ __launch_bounds__(256) void gemm_wmma_f16(
    const _Float16* __restrict__ A, int lda, long sA,
    const _Float16* __restrict__ Bt, int ldb, long sB,
    void* __restrict__ C, int ldc, long sC,
    const float* __restrict__ bias, float scale, int K) {
  constexpr int KSTEP = 64;
  constexpr int LK = KSTEP + 8;  // pad to spread LDS banks
  __shared__ _Float16 As[2][128 * LK];
  __shared__ _Float16 Bs[2][128 * LK];

  const int z = blockIdx.z;
  A  += (long)z * sA;
  Bt += (long)z * sB;

  const int bm = blockIdx.y * 128;
  const int bn = blockIdx.x * 128;
  const int tid  = threadIdx.x;
  const int lane = tid & 31;
  const int w    = tid >> 5;   // 0..7
  const int wm   = w >> 1;     // 0..3  (32 rows each)
  const int wn   = w & 1;      // 0..1  (64 cols each)

  v8f acc[2][4] = {};

  // cooperative stage of one 128x64 slab pair (16B per lane per chunk)
  auto stage = [&](int buf, int k0) {
#pragma unroll
    for (int i = 0; i < 4; ++i) {
      const int chunk = tid + i * 256;       // 0..1023
      const int row = chunk >> 3;            // 0..127
      const int c8  = (chunk & 7) * 8;       // 0..56
      cp16_to_lds(&A[(long)(bm + row) * lda + (k0 + c8)], &As[buf][row * LK + c8]);
      cp16_to_lds(&Bt[(long)(bn + row) * ldb + (k0 + c8)], &Bs[buf][row * LK + c8]);
    }
  };

  const int nsteps = K / KSTEP;
  stage(0, 0);

  for (int s = 0; s < nsteps; ++s) {
    const int cur = s & 1;
    wait_async_loads();   // our slab-s loads are in LDS
    __syncthreads();      // everyone's are; everyone's slab-(s-1) reads drained
    if (s + 1 < nsteps) stage(cur ^ 1, (s + 1) * KSTEP);  // prefetch next slab

    const _Float16* Ac = As[cur];
    const _Float16* Bc = Bs[cur];
#pragma unroll
    for (int kk = 0; kk < KSTEP; kk += 32) {
      v16h af[2], bf[4];
      const int r = lane & 15;
      // A frag (16x32 f16): lanes 0-15 -> K 0..7 & 16..23; lanes 16-31 -> 8..15 & 24..31
      const int akb = (lane >> 4) * 8;
#pragma unroll
      for (int sm = 0; sm < 2; ++sm) {
        const _Float16* base = &Ac[(wm * 32 + sm * 16 + r) * LK + kk + akb];
        af[sm] = make_v16(*(const v8h*)base, *(const v8h*)(base + 16));
      }
      // B frag (32x16 f16): lane holds column n=lane&15, K 0..7 & 16..23 (lo lanes) / 8..15 & 24..31 (hi lanes)
      const int bkb = (lane >> 4) * 8;
#pragma unroll
      for (int sn = 0; sn < 4; ++sn) {
        const _Float16* base = &Bc[(wn * 64 + sn * 16 + r) * LK + kk + bkb];
        bf[sn] = make_v16(*(const v8h*)base, *(const v8h*)(base + 16));
      }
#pragma unroll
      for (int sm = 0; sm < 2; ++sm)
#pragma unroll
        for (int sn = 0; sn < 4; ++sn)
          acc[sm][sn] = __builtin_amdgcn_wmma_f32_16x16x32_f16(
              false, af[sm], false, bf[sn], (short)0, acc[sm][sn], false, false);
    }
  }

  // Epilogue: C/D layout -> VGPR vi: lanes 0-15 = M=vi, lanes 16-31 = M=8+vi; N=lane&15
  const int ln = lane & 15;
  const int mh = (lane >> 4) * 8;
#pragma unroll
  for (int sm = 0; sm < 2; ++sm) {
#pragma unroll
    for (int sn = 0; sn < 4; ++sn) {
      const int n = bn + wn * 64 + sn * 16 + ln;
      const float bv = HAS_BIAS ? bias[n] : 0.0f;
#pragma unroll
      for (int vi = 0; vi < 8; ++vi) {
        const long m = bm + wm * 32 + sm * 16 + mh + vi;
        const float val = acc[sm][sn][vi] * scale + bv;
        if (OUT_F16)
          ((_Float16*)C)[(long)z * sC + m * ldc + n] = (_Float16)val;
        else
          ((float*)C)[(long)z * sC + m * ldc + n] = val;
      }
    }
  }
}

// f32 -> f16 elementwise
__global__ void cvt_f32_f16(const float* __restrict__ in, _Float16* __restrict__ out, long n) {
  const long i = (long)blockIdx.x * blockDim.x + threadIdx.x;
  if (i < n) out[i] = (_Float16)in[i];
}

// f32 [R x C] -> f16 [C x R] (tiled transpose + convert). R,C multiples of 32.
__global__ void transpose_f32_f16(const float* __restrict__ in, _Float16* __restrict__ out,
                                  int R, int C) {
  __shared__ float tile[32][33];
  const int c0 = blockIdx.x * 32, r0 = blockIdx.y * 32;
  for (int j = threadIdx.y; j < 32; j += 8)
    tile[j][threadIdx.x] = in[(long)(r0 + j) * C + c0 + threadIdx.x];
  __syncthreads();
  for (int j = threadIdx.y; j < 32; j += 8)
    out[(long)(c0 + j) * R + r0 + threadIdx.x] = (_Float16)tile[threadIdx.x][j];
}

// f16 [R x C] (leading dim lda, batched) -> f16 [C x R]
__global__ void transpose_f16(const _Float16* __restrict__ in, int lda, long inB,
                              _Float16* __restrict__ out, long outB, int R, int C) {
  __shared__ _Float16 tile[32][33];
  const int z = blockIdx.z;
  in  += (long)z * inB;
  out += (long)z * outB;
  const int c0 = blockIdx.x * 32, r0 = blockIdx.y * 32;
  for (int j = threadIdx.y; j < 32; j += 8)
    tile[j][threadIdx.x] = in[(long)(r0 + j) * lda + c0 + threadIdx.x];
  __syncthreads();
  for (int j = threadIdx.y; j < 32; j += 8)
    out[(long)(c0 + j) * R + r0 + threadIdx.x] = tile[threadIdx.x][j];
}

// Row softmax over n cols, in-place f32, plus f16 copy for the ctx GEMM.
__global__ __launch_bounds__(256) void softmax_row(float* __restrict__ sc,
                                                   _Float16* __restrict__ ah, int n) {
  const long row = blockIdx.x;
  float* p = sc + row * (long)n;
  _Float16* ph = ah + row * (long)n;
  const int t = threadIdx.x;
  __shared__ float red[8];

  float m = -3.4e38f;
  for (int c = t; c < n; c += 256) m = fmaxf(m, p[c]);
  for (int off = 16; off > 0; off >>= 1) m = fmaxf(m, __shfl_xor(m, off, 32));
  if ((t & 31) == 0) red[t >> 5] = m;
  __syncthreads();
  float m0 = red[0];
#pragma unroll
  for (int i = 1; i < 8; ++i) m0 = fmaxf(m0, red[i]);
  __syncthreads();

  float s = 0.f;
  for (int c = t; c < n; c += 256) {
    const float e = __expf(p[c] - m0);
    p[c] = e;
    s += e;
  }
  for (int off = 16; off > 0; off >>= 1) s += __shfl_xor(s, off, 32);
  if ((t & 31) == 0) red[t >> 5] = s;
  __syncthreads();
  float s0 = 0.f;
#pragma unroll
  for (int i = 0; i < 8; ++i) s0 += red[i];
  const float inv = 1.f / s0;
  for (int c = t; c < n; c += 256) {
    const float a = p[c] * inv;
    p[c] = a;
    ph[c] = (_Float16)a;
  }
}

extern "C" void kernel_launch(void* const* d_in, const int* in_sizes, int n_in,
                              void* d_out, int out_size, void* d_ws, size_t ws_size,
                              hipStream_t stream) {
  using namespace cfg;
  const float* x     = (const float*)d_in[0];  // [B,S,D]
  const float* w_qkv = (const float*)d_in[1];  // [D,3D]
  const float* b_qkv = (const float*)d_in[2];  // [3D]
  const float* w_out = (const float*)d_in[3];  // [D,D]
  const float* b_out = (const float*)d_in[4];  // [D]

  float* out  = (float*)d_out;             // [B,S,D]
  float* attn = out + BS * (long)D;        // [B,S,S] (also used as raw-score scratch)

  // workspace carve-up (all sizes are multiples of 512 KiB -> aligned)
  char* w = (char*)d_ws;
  _Float16* x_h    = (_Float16*)w; w += (size_t)BS * D * 2;    // 8 MiB
  _Float16* ctx_h  = x_h;                                      // reuse after stage 1
  _Float16* wqkvT  = (_Float16*)w; w += (size_t)D3 * D * 2;    // 1.5 MiB
  _Float16* woutT  = (_Float16*)w; w += (size_t)D * D * 2;     // 0.5 MiB
  _Float16* qkv_h  = (_Float16*)w; w += (size_t)BS * D3 * 2;   // 24 MiB
  _Float16* vT     = (_Float16*)w; w += (size_t)B * D * (size_t)S * 2;  // 8 MiB
  _Float16* attn_h = (_Float16*)w; w += (size_t)B * (size_t)S * S * 2;  // 32 MiB
  (void)ws_size; (void)in_sizes; (void)n_in; (void)out_size;

  // 0) convert inputs to f16 (x) and transposed f16 (weights)
  cvt_f32_f16<<<dim3((unsigned)((BS * D) / 256)), dim3(256), 0, stream>>>(x, x_h, BS * (long)D);
  transpose_f32_f16<<<dim3(D3 / 32, D / 32), dim3(32, 8), 0, stream>>>(w_qkv, wqkvT, D, D3);
  transpose_f32_f16<<<dim3(D / 32, D / 32), dim3(32, 8), 0, stream>>>(w_out, woutT, D, D);

  // 1) qkv = x @ w_qkv + b_qkv   -> f16 [BS, 3D]
  gemm_wmma_f16<true, true><<<dim3(D3 / 128, BS / 128, 1), dim3(256), 0, stream>>>(
      x_h, D, 0, wqkvT, D, 0, qkv_h, D3, 0, b_qkv, 1.0f, D);

  // 1b) vT[b] = v[b]^T  ([S,D] slice of qkv at +2D -> [D,S])
  transpose_f16<<<dim3(D / 32, S / 32, B), dim3(32, 8), 0, stream>>>(
      qkv_h + 2 * D, D3, (long)S * D3, vT, (long)D * S, S, D);

  // 2) raw scores = (q @ k^T) / 64  -> f32 into d_out attn region
  gemm_wmma_f16<false, false><<<dim3(S / 128, S / 128, B), dim3(256), 0, stream>>>(
      qkv_h, D3, (long)S * D3,            // q slice
      qkv_h + D, D3, (long)S * D3,        // k slice as Bt
      attn, S, (long)S * S, nullptr, 1.0f / 64.0f, D);

  // 3) softmax rows: f32 in-place in d_out + f16 copy for ctx GEMM
  softmax_row<<<dim3(B * S), dim3(256), 0, stream>>>(attn, attn_h, S);

  // 4) ctx = attn @ v  -> f16 [B,S,D]
  gemm_wmma_f16<true, false><<<dim3(D / 128, S / 128, B), dim3(256), 0, stream>>>(
      attn_h, S, (long)S * S, vT, S, (long)D * S, ctx_h, D, (long)S * D, nullptr, 1.0f, S);

  // 5) out = ctx @ w_out + b_out  -> f32 [B,S,D]
  gemm_wmma_f16<false, true><<<dim3(D / 128, BS / 128, 1), dim3(256), 0, stream>>>(
      ctx_h, D, 0, woutT, D, 0, out, D, 0, b_out, 1.0f, D);
}
